// LlamaFFNNetwork_30545807409458
// MI455X (gfx1250) — compile-verified, matrix operand order checked by reading, not run on hardware
//
#include <hip/hip_runtime.h>
#include <hip/hip_bf16.h>

// Per-expert SwiGLU FFN for MI455X (gfx1250), bf16 WMMA path.
// E=8, T=1024, D=2048, H=5632. Compute-bound => v_wmma_f32_16x16x32_bf16,
// fp32 operands converted to bf16 on the fly, f32 accumulate.
#define E_ 8
#define T_ 1024
#define D_ 2048
#define H_ 5632

#define BM 128   // tile rows (tokens)
#define BN 128   // tile cols (H or D)
#define BK 32    // K step (== WMMA K for bf16)
#define LDA 48   // LDS row stride (elems): 32 data + 16 pad -> 96B rows (32B aligned)
#define LDB 48   // LDS row stride for transposed B tiles

typedef __attribute__((ext_vector_type(16))) __bf16 v16bf;
typedef __attribute__((ext_vector_type(8)))  float  v8f;
typedef __attribute__((ext_vector_type(4)))  unsigned int v4u;
typedef __attribute__((ext_vector_type(8)))  int v8i;
typedef __attribute__((ext_vector_type(4)))  int v4i;

#define HAVE_TDM (__has_builtin(__builtin_amdgcn_tensor_load_to_lds) && \
                  __has_builtin(__builtin_amdgcn_s_wait_tensorcnt))

__device__ __forceinline__ unsigned short f2bf(float f) {
    // round-to-nearest-even fp32 -> bf16
    unsigned u = __float_as_uint(f);
    u += 0x7FFFu + ((u >> 16) & 1u);
    return (unsigned short)(u >> 16);
}

// ---------------------------------------------------------------------------
// Kernel 1: y = silu(x@W1) * (x@W2), y stored bf16 in workspace.
// Block = 256 threads = 8 wave32s, wave grid 2(M)x4(N), each wave 64x32
// output = 4x2 wmma tiles, two accumulator sets (y1, y2) share the A tile.
// ---------------------------------------------------------------------------
__global__ __launch_bounds__(256) void ffn_fc12_kernel(
    const float* __restrict__ x, const float* __restrict__ W1,
    const float* __restrict__ W2, unsigned short* __restrict__ y)
{
    __shared__ __align__(32) unsigned short As [BM * LDA];
    __shared__ __align__(32) unsigned short B1s[BN * LDB];   // transposed [h][k]
    __shared__ __align__(32) unsigned short B2s[BN * LDB];

    const int e  = blockIdx.z;
    const int t0 = blockIdx.y * BM;
    const int h0 = blockIdx.x * BN;

    const float* xe = x  + ((size_t)e * T_ + t0) * D_;
    const float* w1 = W1 + (size_t)e * D_ * H_ + h0;
    const float* w2 = W2 + (size_t)e * D_ * H_ + h0;

    const int tid  = threadIdx.x;
    const int lane = tid & 31;
    const int wave = tid >> 5;
    const int wm   = wave & 1;          // wave row (0..1) -> 64 tokens
    const int wn   = wave >> 1;         // wave col (0..3) -> 32 h
    const int fr   = lane & 15;         // row/col within 16x16 fragment
    const int kb   = (lane >> 4) * 16;  // K half per lane group (0 | 16)

    v8f acc1[4][2] = {};
    v8f acc2[4][2] = {};

    for (int kt = 0; kt < D_ / BK; ++kt) {
        const int k0 = kt * BK;

        // Prefetch next K-tile of the streamed weights (speculative, OOB-safe).
        __builtin_prefetch(w1 + (size_t)(k0 + BK) * H_ + (tid & 127), 0, 1);
        __builtin_prefetch(w2 + (size_t)(k0 + BK) * H_ + (tid & 127), 0, 1);

        // ---- A tile: 128x32 f32 -> bf16 LDS (row-major) -------------------
        #pragma unroll
        for (int i = 0; i < 4; ++i) {
            int idx = tid + i * 256;          // 1024 float4 chunks
            int row = idx >> 3;               // 8 chunks per 32-elem row
            int col = (idx & 7) * 4;
            float4 g = *(const float4*)(xe + (size_t)row * D_ + k0 + col);
            unsigned p0 = (unsigned)f2bf(g.x) | ((unsigned)f2bf(g.y) << 16);
            unsigned p1 = (unsigned)f2bf(g.z) | ((unsigned)f2bf(g.w) << 16);
            *(uint2*)&As[row * LDA + col] = make_uint2(p0, p1);
        }
        // ---- B tiles: 32x128 f32 -> bf16 LDS, stored transposed [h][k] ----
        #pragma unroll
        for (int i = 0; i < 4; ++i) {
            int idx = tid + i * 256;
            int row = idx >> 5;               // k row (0..31), 32 float4 per row
            int col = (idx & 31) * 4;         // h col
            size_t go = (size_t)(k0 + row) * H_ + col;
            float4 g1 = *(const float4*)(w1 + go);
            float4 g2 = *(const float4*)(w2 + go);
            B1s[(col + 0) * LDB + row] = f2bf(g1.x);
            B1s[(col + 1) * LDB + row] = f2bf(g1.y);
            B1s[(col + 2) * LDB + row] = f2bf(g1.z);
            B1s[(col + 3) * LDB + row] = f2bf(g1.w);
            B2s[(col + 0) * LDB + row] = f2bf(g2.x);
            B2s[(col + 1) * LDB + row] = f2bf(g2.y);
            B2s[(col + 2) * LDB + row] = f2bf(g2.z);
            B2s[(col + 3) * LDB + row] = f2bf(g2.w);
        }
        __syncthreads();

        // ---- fragment loads: 16 contiguous bf16 (32B) per lane ------------
        v16bf af[4], b1f[2], b2f[2];
        #pragma unroll
        for (int mi = 0; mi < 4; ++mi)
            af[mi] = *(const v16bf*)&As[(wm * 64 + mi * 16 + fr) * LDA + kb];
        #pragma unroll
        for (int ni = 0; ni < 2; ++ni) {
            b1f[ni] = *(const v16bf*)&B1s[(wn * 32 + ni * 16 + fr) * LDB + kb];
            b2f[ni] = *(const v16bf*)&B2s[(wn * 32 + ni * 16 + fr) * LDB + kb];
        }
        __syncthreads();

        // ---- 16 WMMAs per iteration (A reused across both weight mats) ----
        #pragma unroll
        for (int mi = 0; mi < 4; ++mi)
            #pragma unroll
            for (int ni = 0; ni < 2; ++ni) {
                acc1[mi][ni] = __builtin_amdgcn_wmma_f32_16x16x32_bf16(
                    false, af[mi], false, b1f[ni], (short)0, acc1[mi][ni], false, false);
                acc2[mi][ni] = __builtin_amdgcn_wmma_f32_16x16x32_bf16(
                    false, af[mi], false, b2f[ni], (short)0, acc2[mi][ni], false, false);
            }
    }

    // ---- epilogue: silu(y1) * y2 -> bf16 y -------------------------------
    // C/D layout: VGPR r, lane l -> N = l%16, M = r + (l/16)*8
    const size_t ybase = (size_t)e * T_ * H_;
    #pragma unroll
    for (int mi = 0; mi < 4; ++mi)
        #pragma unroll
        for (int ni = 0; ni < 2; ++ni)
            #pragma unroll
            for (int r = 0; r < 8; ++r) {
                int m = r + (lane >> 4) * 8;
                int n = lane & 15;
                int t = t0 + wm * 64 + mi * 16 + m;
                int h = h0 + wn * 32 + ni * 16 + n;
                float a = acc1[mi][ni][r];
                float b = acc2[mi][ni][r];
                // silu via v_rcp_f32 (avoid the IEEE div chain)
                float s = a * __builtin_amdgcn_rcpf(1.0f + __expf(-a));
                y[ybase + (size_t)t * H_ + h] = f2bf(s * b);
            }
}

// ---------------------------------------------------------------------------
// Kernel 2: out = y @ W3   ([T,H] bf16 x [H,D] f32->bf16 -> [T,D] f32)
// A tile (already bf16) is fetched by the Tensor Data Mover straight into
// LDS, with TDM padding reproducing the 96B LDS row stride.
// ---------------------------------------------------------------------------
__global__ __launch_bounds__(256) void ffn_fc3_kernel(
    const unsigned short* __restrict__ y, const float* __restrict__ W3,
    float* __restrict__ out)
{
    __shared__ __align__(32) unsigned short As[BM * LDA];
    __shared__ __align__(32) unsigned short Bs[BN * LDB];   // transposed [d][k]

    const int e  = blockIdx.z;
    const int t0 = blockIdx.y * BM;
    const int d0 = blockIdx.x * BN;

    const unsigned short* ye = y + ((size_t)e * T_ + t0) * H_;
    const float*          w3 = W3 + (size_t)e * H_ * D_ + d0;

    const int tid  = threadIdx.x;
    const int lane = tid & 31;
    const int wave = tid >> 5;
    const int wm   = wave & 1;
    const int wn   = wave >> 1;
    const int fr   = lane & 15;
    const int kb   = (lane >> 4) * 16;

#if HAVE_TDM
    // LDS byte offset of As: low 32 bits of the generic pointer are the
    // wave-relative LDS address (aperture bits live in the upper half).
    const unsigned lds_a = (unsigned)(size_t)(void*)As;
#endif

    v8f acc[4][2] = {};

    for (int kt = 0; kt < H_ / BK; ++kt) {
        const int k0 = kt * BK;

        __builtin_prefetch(w3 + (size_t)(k0 + BK) * D_ + (tid & 127), 0, 1);

#if HAVE_TDM
        // ---- A tile via TDM: 128 rows x 32 bf16, global stride H ----------
        if (tid < 32) {   // one issuing wave (TDM ignores EXEC; dup-safe)
            size_t ga = (size_t)(const void*)(ye + k0);
            v4u g0 = { 1u,                                   // count=1 (user D#)
                       lds_a,                                // lds_addr
                       (unsigned)ga,                         // global_addr[31:0]
                       ((unsigned)(ga >> 32) & 0x01ffffffu)  // global_addr[56:32]
                       | 0x80000000u };                      // type=2 ("image")
            v8i g1 = { (int)((1u << 16)    // data_size = 2 bytes
                           | (1u << 20)    // pad_enable
                           | (3u << 22)    // pad_interval: 16 DWORDs (64B row)
                           | (7u << 25)),  // pad_amount:    8 DWORDs (32B pad)
                       (int)((unsigned)H_ << 16),                        // tensor_dim0[15:0]
                       (int)(((unsigned)H_ >> 16) | ((unsigned)T_ << 16)),// dim0 hi | dim1 lo
                       (int)(32u << 16),   // dim1 hi=0 | tile_dim0=32
                       128,                // tile_dim1=128 | tile_dim2=0
                       (int)H_,            // tensor_dim0_stride[31:0]
                       0, 0 };
            v4i gz = { 0, 0, 0, 0 };
#if __clang_major__ >= 23
            v8i gz8 = { 0, 0, 0, 0, 0, 0, 0, 0 };
            __builtin_amdgcn_tensor_load_to_lds(g0, g1, gz, gz, gz8, 0);
#else
            __builtin_amdgcn_tensor_load_to_lds(g0, g1, gz, gz, 0);
#endif
        }
#else
        // ---- A tile: 128x32 bf16, straight 16B copies ---------------------
        #pragma unroll
        for (int i = 0; i < 2; ++i) {
            int idx = tid * 2 + i;            // 512 16-byte chunks
            int row = idx >> 2;               // 4 chunks per row
            int c8  = (idx & 3) * 8;
            uint4 g = *(const uint4*)(ye + (size_t)row * H_ + k0 + c8);
            *(uint4*)&As[row * LDA + c8] = g;
        }
#endif
        // ---- B tile: 32x128 f32 -> bf16 LDS transposed [d][k] -------------
        #pragma unroll
        for (int i = 0; i < 4; ++i) {
            int idx = tid + i * 256;
            int row = idx >> 5;
            int col = (idx & 31) * 4;
            float4 g = *(const float4*)(w3 + (size_t)(k0 + row) * D_ + col);
            Bs[(col + 0) * LDB + row] = f2bf(g.x);
            Bs[(col + 1) * LDB + row] = f2bf(g.y);
            Bs[(col + 2) * LDB + row] = f2bf(g.z);
            Bs[(col + 3) * LDB + row] = f2bf(g.w);
        }
#if HAVE_TDM
        if (tid < 32) __builtin_amdgcn_s_wait_tensorcnt(0);  // A tile landed
#endif
        __syncthreads();

        v16bf af[4], bfr[2];
        #pragma unroll
        for (int mi = 0; mi < 4; ++mi)
            af[mi] = *(const v16bf*)&As[(wm * 64 + mi * 16 + fr) * LDA + kb];
        #pragma unroll
        for (int ni = 0; ni < 2; ++ni)
            bfr[ni] = *(const v16bf*)&Bs[(wn * 32 + ni * 16 + fr) * LDB + kb];
        __syncthreads();

        #pragma unroll
        for (int mi = 0; mi < 4; ++mi)
            #pragma unroll
            for (int ni = 0; ni < 2; ++ni)
                acc[mi][ni] = __builtin_amdgcn_wmma_f32_16x16x32_bf16(
                    false, af[mi], false, bfr[ni], (short)0, acc[mi][ni], false, false);
    }

    const size_t obase = (size_t)e * T_ * D_;
    #pragma unroll
    for (int mi = 0; mi < 4; ++mi)
        #pragma unroll
        for (int ni = 0; ni < 2; ++ni)
            #pragma unroll
            for (int r = 0; r < 8; ++r) {
                int m = r + (lane >> 4) * 8;
                int n = lane & 15;
                int t = t0 + wm * 64 + mi * 16 + m;
                int d = d0 + wn * 32 + ni * 16 + n;
                out[obase + (size_t)t * D_ + d] = acc[mi][ni][r];
            }
}

// ---------------------------------------------------------------------------
extern "C" void kernel_launch(void* const* d_in, const int* in_sizes, int n_in,
                              void* d_out, int out_size, void* d_ws, size_t ws_size,
                              hipStream_t stream) {
    const float* x  = (const float*)d_in[0];   // [E,T,D]
    const float* W1 = (const float*)d_in[1];   // [E,D,H]
    const float* W2 = (const float*)d_in[2];   // [E,D,H]
    const float* W3 = (const float*)d_in[3];   // [E,H,D]
    float* out = (float*)d_out;                // [E,T,D] f32

    // Intermediate y = silu(x@W1)*(x@W2): E*T*H bf16 = ~92.3 MB in workspace.
    unsigned short* yws = (unsigned short*)d_ws;

    dim3 blk(256);
    dim3 g1(H_ / BN, T_ / BM, E_);   // 44 x 8 x 8
    hipLaunchKernelGGL(ffn_fc12_kernel, g1, blk, 0, stream, x, W1, W2, yws);

    dim3 g2(D_ / BN, T_ / BM, E_);   // 16 x 8 x 8
    hipLaunchKernelGGL(ffn_fc3_kernel, g2, blk, 0, stream, yws, W3, out);
}
